// QuantizedConv2d_45191645889184
// MI455X (gfx1250) — 8-bit matrix operand order repaired and checked by reading, compiled, not run on hardware
//
#include <hip/hip_runtime.h>


typedef int v8i __attribute__((ext_vector_type(8)));

#define C_IN   128
#define C_OUT  256
#define HDIM   56
#define WDIM   56
#define HW     3136
#define NBATCH 8
#define NPIX   25088
#define KTOT   1152
#define KBLK   18
#define MT     16
#define NT     1568
#define NGRP   392
#define WFRAG_DW (MT*KBLK*32*8)
#define CFRAG_DW (NT*KBLK*32*8)
#define XB_DW    (NPIX*C_IN/4)

__device__ __forceinline__ int ka_off(int v, int h) {
    return (v >> 2) * 32 + h * 16 + (v & 3) * 4;
}
__device__ __forceinline__ int kb_off(int v, int h) {
    return (v >> 2) * 32 + h * 16 + (v & 3) * 4;
}

__global__ void __launch_bounds__(256)
quantize_x(const int* __restrict__ x, unsigned* __restrict__ xb) {
    int tid = blockIdx.x * 256 + threadIdx.x;
    if (tid >= XB_DW) return;
    int4 q = reinterpret_cast<const int4*>(x)[tid];
    unsigned d = (unsigned)(q.x & 0xFF)
               | ((unsigned)(q.y & 0xFF) << 8)
               | ((unsigned)(q.z & 0xFF) << 16)
               | ((unsigned)(q.w & 0xFF) << 24);
    xb[tid] = d;
}

__global__ void __launch_bounds__(256)
pack_weights(const int* __restrict__ w, unsigned* __restrict__ wlo,
             unsigned* __restrict__ whi) {
    int tid = blockIdx.x * 256 + threadIdx.x;
    if (tid >= WFRAG_DW) return;
    int v    = tid & 7;
    int lane = (tid >> 3) & 31;
    int blk  = tid >> 8;
    int kb   = blk % KBLK;
    int mt   = blk / KBLK;
    int co   = mt * 16 + (lane & 15);
    int half = lane >> 4;
    int kbase = kb * 64 + ka_off(v, half);
    int4 q = *reinterpret_cast<const int4*>(w + co * KTOT + kbase);
    unsigned lo = (unsigned)(q.x & 0xFF)
                | ((unsigned)(q.y & 0xFF) << 8)
                | ((unsigned)(q.z & 0xFF) << 16)
                | ((unsigned)(q.w & 0xFF) << 24);
    unsigned hi = (unsigned)((q.x >> 8) & 0xFF)
                | ((unsigned)((q.y >> 8) & 0xFF) << 8)
                | ((unsigned)((q.z >> 8) & 0xFF) << 16)
                | ((unsigned)((q.w >> 8) & 0xFF) << 24);
    wlo[tid] = lo;
    whi[tid] = hi;
}

__global__ void __launch_bounds__(256)
pack_cols(const unsigned char* __restrict__ xb, unsigned* __restrict__ cf) {
    int tid = blockIdx.x * 256 + threadIdx.x;
    if (tid >= CFRAG_DW) return;
    int v    = tid & 7;
    int lane = (tid >> 3) & 31;
    int blk  = tid >> 8;
    int kb   = blk % KBLK;
    int nt   = blk / KBLK;
    int pix  = nt * 16 + (lane & 15);
    int half = lane >> 4;
    int kbase = kb * 64 + kb_off(v, half);
    int n  = pix / HW;
    int hw = pix - n * HW;
    int h  = hw / WDIM;
    int ww = hw - h * WDIM;
    unsigned d = 0;
#pragma unroll
    for (int i = 0; i < 4; ++i) {
        int K  = kbase + i;
        int ci = K / 9;
        int r  = K - ci * 9;
        int kh = r / 3, kw = r - kh * 3;
        int y  = h + kh - 1;
        int xx = ww + kw - 1;
        unsigned b = 0;
        if ((unsigned)y < (unsigned)HDIM && (unsigned)xx < (unsigned)WDIM)
            b = xb[((n * C_IN + ci) * HDIM + y) * WDIM + xx];
        d |= b << (8 * i);
    }
    cf[tid] = d;
}

__global__ void __launch_bounds__(256)
qconv_wmma(const unsigned* __restrict__ wlo, const unsigned* __restrict__ whi,
           const unsigned* __restrict__ cf, const int* __restrict__ bias,
           const int* __restrict__ shift_p, float* __restrict__ out) {
    int lane = threadIdx.x & 31;
    int wave = threadIdx.x >> 5;
    int wid  = blockIdx.x * 8 + wave;
    if (wid >= MT * NGRP) return;
    int mt = wid / NGRP;
    int ng = wid - mt * NGRP;

    v8i acc_lo[4] = {};
    v8i acc_hi[4] = {};

    const unsigned a_base = ((unsigned)(mt * KBLK) * 32u + (unsigned)lane) * 8u;
    const unsigned b_row  = ((unsigned)(ng * 4) * KBLK * 32u) * 8u;

    for (int kb = 0; kb < KBLK; ++kb) {
        v8i a_lo = *reinterpret_cast<const v8i*>(wlo + a_base + (unsigned)kb * 256u);
        v8i a_hi = *reinterpret_cast<const v8i*>(whi + a_base + (unsigned)kb * 256u);

        if (kb + 1 < KBLK) {
#pragma unroll
            for (int t = 0; t < 4; ++t)
                __builtin_prefetch(
                    cf + b_row + ((unsigned)(t * KBLK + kb + 1) * 32u + (unsigned)lane) * 8u,
                    0, 1);
        }

#pragma unroll
        for (int t = 0; t < 4; ++t) {
            unsigned off = b_row + ((unsigned)(t * KBLK + kb) * 32u + (unsigned)lane) * 8u;
            v8i b = *reinterpret_cast<const v8i*>(cf + off);
            acc_hi[t] = __builtin_amdgcn_wmma_i32_16x16x64_iu8(
                true,  a_hi, true, b, acc_hi[t], false, false);
            acc_lo[t] = __builtin_amdgcn_wmma_i32_16x16x64_iu8(
                false, a_lo, true, b, acc_lo[t], false, false);
        }
    }

    int shift = shift_p[0];
    int half  = lane >> 4;
    int ln    = lane & 15;
#pragma unroll
    for (int t = 0; t < 4; ++t) {
        int pix = ng * 64 + t * 16 + ln;
        int n   = pix / HW;
        int hw  = pix - n * HW;
#pragma unroll
        for (int r = 0; r < 8; ++r) {
            int co = mt * 16 + half * 8 + r;
            long long acc = ((long long)acc_hi[t][r] << 8)
                          + (long long)acc_lo[t][r]
                          + (long long)bias[co];
            long long res = acc >> shift;
            out[(size_t)n * (C_OUT * HW) + (size_t)co * HW + hw] = (float)res;
        }
    }
}

extern "C" void kernel_launch(void* const* d_in, const int* in_sizes, int n_in,
                              void* d_out, int out_size, void* d_ws, size_t ws_size,
                              hipStream_t stream) {
    const int* x     = (const int*)d_in[0];
    const int* w     = (const int*)d_in[1];
    const int* bias  = (const int*)d_in[2];
    const int* shift = (const int*)d_in[3];
    float* out = (float*)d_out;

    unsigned* wlo = (unsigned*)d_ws;
    unsigned* whi = wlo + WFRAG_DW;
    unsigned* cf  = whi + WFRAG_DW;
    unsigned* xb  = cf  + CFRAG_DW;

    quantize_x  <<<(XB_DW    + 255) / 256, 256, 0, stream>>>(x, xb);
    pack_weights<<<(WFRAG_DW + 255) / 256, 256, 0, stream>>>(w, wlo, whi);
    pack_cols   <<<(CFRAG_DW + 255) / 256, 256, 0, stream>>>(
        (const unsigned char*)xb, cf);

    int jobs = MT * NGRP;
    qconv_wmma<<<(jobs + 7) / 8, 256, 0, stream>>>(wlo, whi, cf, bias, shift, out);
}
